// EdgeNetwork_34875134443557
// MI455X (gfx1250) — compile-verified, matrix operand order checked by reading, not run on hardware
//
#include <hip/hip_runtime.h>

typedef __attribute__((ext_vector_type(16))) _Float16 v16h;
typedef __attribute__((ext_vector_type(2)))  _Float16 h2;
typedef __attribute__((ext_vector_type(8)))  float    v8f;

#define N_ATOMS   50000
#define N_EDGES   100000
#define ATOM_DIM  64
#define BOND_DIM  16
#define UNITS     4096
#define KTOT      1088            // (16 kernel rows + 1 bias row) * 64
#define NCHUNKS   34              // KTOT / 32
#define NTILES    (N_EDGES / 16)  // 6250 edge tiles of 16
#define WAVES_PER_BLOCK 4
#define BLOCK_THREADS   128
#define GRID_BLOCKS     256
#define WFRAG_H2  (NCHUNKS * 4 * 32 * 8)  // 34816 h2 entries
#define WLDS_BYTES (WFRAG_H2 * 4)         // 139264 bytes LDS

// K offset within a 32-wide chunk for the 16-bit A/B fragment layout
// (ISA 7.12.2): lanes 0-15 hold K {0..7,16..23}, lanes 16-31 hold {8..15,24..31}
__device__ __forceinline__ int fragK(int v, int group) {
    return (v < 4) ? (2 * v + 8 * group) : (16 + 2 * (v - 4) + 8 * group);
}

__global__ void edge_zero_kernel(float* __restrict__ out, int n4) {
    int i = blockIdx.x * blockDim.x + threadIdx.x;
    if (i < n4) {
        float4 z = make_float4(0.f, 0.f, 0.f, 0.f);
        *(float4*)(out + (size_t)i * 4) = z;
    }
}

__global__ void __launch_bounds__(BLOCK_THREADS)
edge_network_kernel(const float* __restrict__ atom,
                    const float* __restrict__ bond,
                    const int*   __restrict__ pair,
                    const float* __restrict__ knl,
                    const float* __restrict__ bias,
                    float* __restrict__ out) {
    extern __shared__ h2 wlds[];  // W fragments, fragment-major: [kk][nt][lane][v]

    // ---- Stage W = [kernel rows 0..15 ; bias] as f16 B-fragments into LDS ----
    for (int idx = threadIdx.x; idx < WFRAG_H2; idx += BLOCK_THREADS) {
        int v    = idx & 7;
        int lane = (idx >> 3) & 31;
        int nt   = (idx >> 8) & 3;
        int kk   = idx >> 10;
        int group = lane >> 4;
        int N     = nt * 16 + (lane & 15);
        int K0    = kk * 32 + fragK(v, group);
        float w0, w1;
        {
            int k = K0 >> 6, j = K0 & 63;
            w0 = (k < 16) ? knl[k * UNITS + N * 64 + j] : bias[N * 64 + j];
        }
        {
            int K1 = K0 + 1;
            int k = K1 >> 6, j = K1 & 63;
            w1 = (k < 16) ? knl[k * UNITS + N * 64 + j] : bias[N * 64 + j];
        }
        wlds[idx] = (h2){(_Float16)w0, (_Float16)w1};
    }
    __syncthreads();

    const int lane   = threadIdx.x & 31;
    const int waveId = threadIdx.x >> 5;
    const int m      = lane & 15;
    const int group  = lane >> 4;

    for (int tile = blockIdx.x * WAVES_PER_BLOCK + waveId; tile < NTILES;
         tile += gridDim.x * WAVES_PER_BLOCK) {
        const int edge = tile * 16 + m;
        const int src  = pair[2 * edge + 1];

        // ---- Load neighbor row into A-fragment layout (two j-halves) ----
        const float* arow = atom + (size_t)src * ATOM_DIM;
        union { v16h v; h2 h[8]; } nf0, nf1;
        #pragma unroll
        for (int v = 0; v < 8; ++v) {
            int jb = fragK(v, group);
            float2 x0 = *(const float2*)(arow + jb);
            float2 x1 = *(const float2*)(arow + 32 + jb);
            nf0.h[v] = (h2){(_Float16)x0.x, (_Float16)x0.y};
            nf1.h[v] = (h2){(_Float16)x1.x, (_Float16)x1.y};
        }

        // ---- Load bond features for this lane's edge (+1.0 bias row) ----
        const float* brow = bond + (size_t)edge * BOND_DIM;
        float bv[17];
        #pragma unroll
        for (int k = 0; k < 16; ++k) bv[k] = brow[k];
        bv[16] = 1.0f;

        v8f c0 = {}, c1 = {}, c2 = {}, c3 = {};

        // ---- K loop: Z-fragment = b[e,k] * nfrag, 4 WMMA per 32-chunk ----
        #pragma unroll
        for (int k = 0; k < 17; ++k) {
            h2 bb = (h2){(_Float16)bv[k], (_Float16)bv[k]};
            #pragma unroll
            for (int jh = 0; jh < 2; ++jh) {
                int kk = k * 2 + jh;
                union { v16h v; h2 h[8]; } af;
                #pragma unroll
                for (int v = 0; v < 8; ++v)
                    af.h[v] = bb * (jh == 0 ? nf0.h[v] : nf1.h[v]);

                const h2* wp = wlds + kk * 1024 + lane * 8;
                v16h wb0 = *(const v16h*)(wp + 0 * 256);
                v16h wb1 = *(const v16h*)(wp + 1 * 256);
                v16h wb2 = *(const v16h*)(wp + 2 * 256);
                v16h wb3 = *(const v16h*)(wp + 3 * 256);
                c0 = __builtin_amdgcn_wmma_f32_16x16x32_f16(false, af.v, false, wb0, (short)0, c0, false, false);
                c1 = __builtin_amdgcn_wmma_f32_16x16x32_f16(false, af.v, false, wb1, (short)0, c1, false, false);
                c2 = __builtin_amdgcn_wmma_f32_16x16x32_f16(false, af.v, false, wb2, (short)0, c2, false, false);
                c3 = __builtin_amdgcn_wmma_f32_16x16x32_f16(false, af.v, false, wb3, (short)0, c3, false, false);
            }
        }

        // ---- Scatter-add: C element (r, lane) -> edge tile*16 + r + 8*group,
        //      feature nt*16 + (lane&15) ----
        const int* pbase = pair + 2 * (tile * 16);
        #pragma unroll
        for (int r = 0; r < 8; ++r) {
            int em  = r + 8 * group;
            int dst = pbase[2 * em];
            float* orow = out + (size_t)dst * ATOM_DIM + m;
            __hip_atomic_fetch_add(orow + 0,  c0[r], __ATOMIC_RELAXED, __HIP_MEMORY_SCOPE_AGENT);
            __hip_atomic_fetch_add(orow + 16, c1[r], __ATOMIC_RELAXED, __HIP_MEMORY_SCOPE_AGENT);
            __hip_atomic_fetch_add(orow + 32, c2[r], __ATOMIC_RELAXED, __HIP_MEMORY_SCOPE_AGENT);
            __hip_atomic_fetch_add(orow + 48, c3[r], __ATOMIC_RELAXED, __HIP_MEMORY_SCOPE_AGENT);
        }
    }
}

extern "C" void kernel_launch(void* const* d_in, const int* in_sizes, int n_in,
                              void* d_out, int out_size, void* d_ws, size_t ws_size,
                              hipStream_t stream) {
    const float* atom = (const float*)d_in[0];
    const float* bond = (const float*)d_in[1];
    const int*   pair = (const int*)d_in[2];
    const float* knl  = (const float*)d_in[3];
    const float* bias = (const float*)d_in[4];
    float* out = (float*)d_out;

    // Allow >64KB dynamic LDS (320KB per WGP on gfx1250). Idempotent; called
    // unconditionally so kernel_launch stays deterministic (no static guards).
    hipFuncSetAttribute((const void*)edge_network_kernel,
                        hipFuncAttributeMaxDynamicSharedMemorySize, WLDS_BYTES);

    int n4 = out_size / 4;  // 3,200,000 floats, divisible by 4
    edge_zero_kernel<<<(n4 + 255) / 256, 256, 0, stream>>>(out, n4);
    edge_network_kernel<<<GRID_BLOCKS, BLOCK_THREADS, WLDS_BYTES, stream>>>(
        atom, bond, pair, knl, bias, out);
}
